// KAN_80736795230878
// MI455X (gfx1250) — matrix operand order repaired and checked by reading, not compiled, not run on hardware
//
#include <hip/hip_runtime.h>
#include <hip/hip_bf16.h>

#define N_ROWS  2048
#define D_IN    64
#define INNER   32
#define HIDDEN  2048
#define CLASSES 10
#define PAIRS   2016
#define KDIM    (PAIRS * INNER)   // 64512

#define BM  256      // output rows per block
#define BN  64       // output cols per block
#define BK  32       // K per step == INNER (one pair unit per step)
#define LDA 40       // At row stride in ushorts (80B -> conflict-free frag reads)
#define LDB 40       // Bt row stride in ushorts
#define XLD 65       // xs row stride in floats (conflict-free dynamic index)

typedef __attribute__((ext_vector_type(16))) __bf16 v16bf;
typedef __attribute__((ext_vector_type(8)))  float  v8f;

union FragBF { v16bf v; uint4 q[2]; };

__device__ __forceinline__ unsigned short f2bf(float f) {
  unsigned int u = __float_as_uint(f);
  u += 0x7FFFu + ((u >> 16) & 1u);      // round-to-nearest-even
  return (unsigned short)(u >> 16);
}

__global__ __launch_bounds__(256)
void fused_pair_gemm_kernel(const float* __restrict__ x,
                            const float* __restrict__ Wu,
                            const float* __restrict__ bu,
                            const float* __restrict__ Wv,
                            const float* __restrict__ bv,
                            float* __restrict__ V) {
  __shared__ __align__(16) float          xs[BM * XLD];     // 65.0 KB
  __shared__ __align__(16) unsigned short At[BM * LDA];     // 20.0 KB
  __shared__ __align__(16) unsigned short Bt[BN * LDB];     //  5.0 KB
  __shared__ __align__(16) float          wu_s[3 * INNER];  // Wu0 | Wu1 | bu

  const int tid  = threadIdx.x;
  const int lane = tid & 31;
  const int wave = tid >> 5;             // 0..7 : 32 rows each
  const int lrow = lane & 15;
  const int hi   = lane >> 4;            // half-wave select (WMMA fragment split)
  const int rowBase = blockIdx.y * BM;
  const int colBase = blockIdx.x * BN;

  // stage this thread's x row (dynamic pair indexing needs LDS, same-thread only)
  {
    const float* src = x + (size_t)(rowBase + tid) * D_IN;
    float* dst = xs + tid * XLD;
    #pragma unroll
    for (int c = 0; c < D_IN; ++c) dst[c] = src[c];
  }
  __syncthreads();

  v8f acc[2][4] = {};

  // B-tile load mapping: thread -> (k row, 8 consecutive cols)
  const int bk = tid >> 3;               // 0..31
  const int bc = (tid & 7) * 8;          // 0..56
  const float* wvp = Wv + (size_t)bk * HIDDEN + colBase + bc;

  int pa = 0, pb = 1;                    // triu pair (i,j), row-major order
  for (int p = 0; p < PAIRS; ++p) {
    // ---- global loads for this step (before barrier, hide latency) ----
    float4 b0 = *(const float4*)(wvp);
    float4 b1 = *(const float4*)(wvp + 4);
    wvp += (size_t)BK * HIDDEN;
    __builtin_prefetch((const void*)wvp, 0, 0);   // global_prefetch next Wv tile
    float wuv = 0.0f;
    if (tid < 96)
      wuv = (tid < 64) ? Wu[(size_t)p * 64 + tid] : bu[(size_t)p * 32 + (tid - 64)];
    float xa = xs[tid * XLD + pa];
    float xb = xs[tid * XLD + pb];

    __syncthreads();   // previous iteration's fragment reads complete

    // ---- store Wv tile transposed to LDS as bf16: Bt[col][k] ----
    {
      unsigned short* bs = Bt + bc * LDB + bk;
      bs[0 * LDB] = f2bf(b0.x); bs[1 * LDB] = f2bf(b0.y);
      bs[2 * LDB] = f2bf(b0.z); bs[3 * LDB] = f2bf(b0.w);
      bs[4 * LDB] = f2bf(b1.x); bs[5 * LDB] = f2bf(b1.y);
      bs[6 * LDB] = f2bf(b1.z); bs[7 * LDB] = f2bf(b1.w);
    }
    if (tid < 96) wu_s[tid] = wuv;
    __syncthreads();   // wu_s ready

    // ---- fused pairwise unit: u[row, 0..31] = relu(xa*Wu0 + xb*Wu1 + bu) ----
    {
      unsigned short* ar = At + tid * LDA;
      #pragma unroll
      for (int i = 0; i < INNER; i += 2) {
        float r0 = fmaf(xa, wu_s[i],     fmaf(xb, wu_s[INNER + i],     wu_s[2 * INNER + i]));
        float r1 = fmaf(xa, wu_s[i + 1], fmaf(xb, wu_s[INNER + i + 1], wu_s[2 * INNER + i + 1]));
        r0 = fmaxf(r0, 0.0f);
        r1 = fmaxf(r1, 0.0f);
        unsigned int pk = (unsigned int)f2bf(r0) | ((unsigned int)f2bf(r1) << 16);
        *(unsigned int*)(ar + i) = pk;
      }
    }
    __syncthreads();   // At ready

    // ---- WMMA: 2 row tiles x 4 col tiles of 16x16, K=32 ----
    FragBF fb[4];
    #pragma unroll
    for (int tc = 0; tc < 4; ++tc) {
      // B 32x16 bf16, this code's k order (same as A): lanes 0-15 hold K0-15, lanes 16-31 hold K16-31
      const unsigned short* bp = Bt + (tc * 16 + lrow) * LDB + hi * 16;
      fb[tc].q[0] = *(const uint4*)(bp);
      fb[tc].q[1] = *(const uint4*)(bp + 8);
    }
    #pragma unroll
    for (int tm = 0; tm < 2; ++tm) {
      // A 16x32 bf16, this code's k order (same as B): lanes 0-15 hold K0-15, lanes 16-31 hold K16-31
      FragBF fa;
      const unsigned short* ap = At + (wave * 32 + tm * 16 + lrow) * LDA + hi * 16;
      fa.q[0] = *(const uint4*)(ap);
      fa.q[1] = *(const uint4*)(ap + 8);
      #pragma unroll
      for (int tc = 0; tc < 4; ++tc) {
        acc[tm][tc] = __builtin_amdgcn_wmma_f32_16x16x32_bf16(
            false, fa.v, false, fb[tc].v, (short)0, acc[tm][tc], false, false);
      }
    }

    // next pair in triu row-major order
    ++pb;
    if (pb == D_IN) { ++pa; pb = pa + 1; }
  }

  // ---- epilogue: v = relu(acc + bv) ----
  #pragma unroll
  for (int tc = 0; tc < 4; ++tc) {
    const int col = colBase + tc * 16 + lrow;
    const float bvv = bv[col];
    #pragma unroll
    for (int tm = 0; tm < 2; ++tm) {
      const int row0 = rowBase + wave * 32 + tm * 16 + hi * 8;
      #pragma unroll
      for (int g = 0; g < 8; ++g) {
        float r = acc[tm][tc][g] + bvv;
        V[(size_t)(row0 + g) * HIDDEN + col] = fmaxf(r, 0.0f);
      }
    }
  }
}

__global__ __launch_bounds__(256)
void head_kernel(const float* __restrict__ V, const float* __restrict__ Wo,
                 const float* __restrict__ bo, float* __restrict__ out) {
  int t = blockIdx.x * 256 + threadIdx.x;
  if (t >= N_ROWS * CLASSES) return;
  int n = t / CLASSES;
  int c = t - n * CLASSES;
  const float* vr = V + (size_t)n * HIDDEN;
  float acc = bo[c];
  for (int h = 0; h < HIDDEN; h += 4) {
    float4 v4 = *(const float4*)(vr + h);
    acc = fmaf(v4.x, Wo[(h + 0) * CLASSES + c], acc);
    acc = fmaf(v4.y, Wo[(h + 1) * CLASSES + c], acc);
    acc = fmaf(v4.z, Wo[(h + 2) * CLASSES + c], acc);
    acc = fmaf(v4.w, Wo[(h + 3) * CLASSES + c], acc);
  }
  out[t] = acc;
}

extern "C" void kernel_launch(void* const* d_in, const int* in_sizes, int n_in,
                              void* d_out, int out_size, void* d_ws, size_t ws_size,
                              hipStream_t stream) {
  (void)in_sizes; (void)n_in; (void)out_size; (void)ws_size;
  const float* x  = (const float*)d_in[0];
  const float* Wu = (const float*)d_in[1];
  const float* bu = (const float*)d_in[2];
  const float* Wv = (const float*)d_in[3];
  const float* bv = (const float*)d_in[4];
  const float* Wo = (const float*)d_in[5];
  const float* bo = (const float*)d_in[6];

  float* V = (float*)d_ws;                     // [N_ROWS, HIDDEN] f32 = 16 MiB scratch
  float* out = (float*)d_out;                  // [N_ROWS, CLASSES] f32

  dim3 grid(HIDDEN / BN, N_ROWS / BM);         // (32, 8)
  fused_pair_gemm_kernel<<<grid, 256, 0, stream>>>(x, Wu, bu, Wv, bv, V);

  int total = N_ROWS * CLASSES;
  head_kernel<<<(total + 255) / 256, 256, 0, stream>>>(V, Wo, bo, out);
}
